// RNNModel_30777735644097
// MI455X (gfx1250) — compile-verified, matrix operand order checked by reading, not run on hardware
//
#include <hip/hip_runtime.h>
#include <hip/hip_bf16.h>
#include <math.h>

// RNN LM forward for MI455X (gfx1250, wave32, WMMA).
//   [gather+proj GEMM] -> [persistent 128-step scan, W_hh in 264KB LDS]
//   -> [pre-split hs/dec_W to bf16 hi+lo planes]
//   -> [decoder GEMM 4096x32000: async global->LDS staging + 3x split-bf16 WMMA]
//
// Workspace layout (fast path, ~175 MiB; falls back to 32 MiB layout if small):
//   [0,16MB)        xin  : [128*32,1024] f32
//   [16MB,32MB)     hs   : [128,32,1024] f32
//   [32MB,+512)     cnt  : 128 int barrier counters
//   [34MB..)        hsH/hsL (8MB each), decWH/decWL (62.5MB each)

typedef __attribute__((ext_vector_type(16))) __bf16 v16bf;
typedef __attribute__((ext_vector_type(8)))  float  v8f;

__device__ __forceinline__ v8f wmma_bf16(v16bf a, v16bf b, v8f c) {
  return __builtin_amdgcn_wmma_f32_16x16x32_bf16(false, a, false, b, (short)0, c,
                                                 false, false);
}

// ISA 16-bit A/B fragment layout: lane half (khalf = lane>>4) owns two
// contiguous 8-element K runs at khalf*8 and 16+khalf*8.
__device__ __forceinline__ v16bf frag_lds(const __bf16* row, int khalf) {
  v16bf f;
  const __bf16* p = row + (khalf << 3);
#pragma unroll
  for (int i = 0; i < 8; ++i) f[i] = p[i];
#pragma unroll
  for (int i = 0; i < 8; ++i) f[i + 8] = p[16 + i];
  return f;
}

__device__ __forceinline__ void split2(float f, __bf16& h, __bf16& l) {
  h = (__bf16)f;
  l = (__bf16)(f - (float)h);
}

// CDNA5 async copy: 16B global -> LDS, tracked by ASYNCcnt (no VGPR round-trip).
__device__ __forceinline__ void async_copy16(unsigned lds_off, const void* gptr) {
  unsigned long long ga = (unsigned long long)(uintptr_t)gptr;
  asm volatile("global_load_async_to_lds_b128 %0, %1, off"
               :: "v"(lds_off), "v"(ga) : "memory");
}
__device__ __forceinline__ void wait_async0() {
  asm volatile("s_wait_asynccnt 0" ::: "memory");
}

constexpr int BM = 64, BN = 128, BK = 32;
constexpr int LDA = BK + 8;  // 40 halfs = 80B row stride (bank-spread, 16B aligned)
constexpr int LDB = BK + 8;

// ---------------------------------------------------------------------------
// f32-input GEMM (fused split conversion): C[M,N] = A[M,K]*B[N,K]^T + bias.
// Used for the encoder (with gather) and as decoder fallback.
// ---------------------------------------------------------------------------
template <bool GATHER>
__global__ __launch_bounds__(256) void gemm_bias_kernel(
    const float* __restrict__ Abase, const int* __restrict__ tok,
    const float* __restrict__ Bmat, const float* __restrict__ bias,
    float* __restrict__ C, int N, int K) {
  __shared__ __attribute__((aligned(16))) __bf16 sAh[BM * LDA];
  __shared__ __attribute__((aligned(16))) __bf16 sAl[BM * LDA];
  __shared__ __attribute__((aligned(16))) __bf16 sBh[BN * LDB];
  __shared__ __attribute__((aligned(16))) __bf16 sBl[BN * LDB];

  const int tid = threadIdx.x;
  const int m0 = blockIdx.y * BM;
  const int n0 = blockIdx.x * BN;
  const int lane = tid & 31, wv = tid >> 5;
  const int khalf = lane >> 4, lrow = lane & 15;
  const int wm = wv >> 2, wn = wv & 3;  // 2 x 4 waves over 64 x 128 tile

  v8f acc[2][2];
#pragma unroll
  for (int a = 0; a < 2; ++a)
#pragma unroll
    for (int b = 0; b < 2; ++b)
#pragma unroll
      for (int r = 0; r < 8; ++r) acc[a][b][r] = 0.0f;

  for (int kk = 0; kk < K; kk += BK) {
#pragma unroll
    for (int j = 0; j < 2; ++j) {  // A: 64x32 f32 -> split bf16
      int idx = tid + 256 * j;
      int row = idx >> 3, kq = idx & 7;
      const float* src;
      if constexpr (GATHER) {
        int t = tok[m0 + row];
        src = Abase + (size_t)t * K + kk + 4 * kq;
      } else {
        src = Abase + (size_t)(m0 + row) * K + kk + 4 * kq;
      }
      float4 v = *(const float4*)src;
      __bf16* ph = &sAh[row * LDA + 4 * kq];
      __bf16* pl = &sAl[row * LDA + 4 * kq];
      split2(v.x, ph[0], pl[0]); split2(v.y, ph[1], pl[1]);
      split2(v.z, ph[2], pl[2]); split2(v.w, ph[3], pl[3]);
    }
#pragma unroll
    for (int j = 0; j < 4; ++j) {  // B: 128x32 f32 -> split bf16
      int idx = tid + 256 * j;
      int row = idx >> 3, kq = idx & 7;
      float4 v = *(const float4*)(Bmat + (size_t)(n0 + row) * K + kk + 4 * kq);
      __bf16* ph = &sBh[row * LDB + 4 * kq];
      __bf16* pl = &sBl[row * LDB + 4 * kq];
      split2(v.x, ph[0], pl[0]); split2(v.y, ph[1], pl[1]);
      split2(v.z, ph[2], pl[2]); split2(v.w, ph[3], pl[3]);
    }
    if (kk + BK < K)
      __builtin_prefetch(Bmat + (size_t)(n0 + (tid >> 1)) * K + kk + BK, 0, 1);
    __syncthreads();

    v16bf bh[2], bl[2];
#pragma unroll
    for (int nt = 0; nt < 2; ++nt) {
      int brow = (32 * wn + 16 * nt + lrow);
      bh[nt] = frag_lds(&sBh[brow * LDB], khalf);
      bl[nt] = frag_lds(&sBl[brow * LDB], khalf);
    }
#pragma unroll
    for (int mt = 0; mt < 2; ++mt) {
      int arow = (32 * wm + 16 * mt + lrow);
      v16bf ah = frag_lds(&sAh[arow * LDA], khalf);
      v16bf al = frag_lds(&sAl[arow * LDA], khalf);
#pragma unroll
      for (int nt = 0; nt < 2; ++nt) {
        acc[mt][nt] = wmma_bf16(ah, bh[nt], acc[mt][nt]);
        acc[mt][nt] = wmma_bf16(ah, bl[nt], acc[mt][nt]);
        acc[mt][nt] = wmma_bf16(al, bh[nt], acc[mt][nt]);
      }
    }
    __syncthreads();
  }

#pragma unroll
  for (int mt = 0; mt < 2; ++mt)
#pragma unroll
    for (int nt = 0; nt < 2; ++nt) {
      int n = n0 + 32 * wn + 16 * nt + lrow;
      float bv = bias[n];
#pragma unroll
      for (int r = 0; r < 8; ++r) {
        int m = m0 + 32 * wm + 16 * mt + 8 * khalf + r;
        C[(size_t)m * N + n] = acc[mt][nt][r] + bv;
      }
    }
}

// ---------------------------------------------------------------------------
// bf16-plane GEMM: A/B pre-split to hi/lo bf16 planes; staging is pure
// async global->LDS (no conversion VALU in the hot loop).
// ---------------------------------------------------------------------------
__global__ __launch_bounds__(256) void gemm_bias_bf16_kernel(
    const __bf16* __restrict__ Ah, const __bf16* __restrict__ Al,
    const __bf16* __restrict__ Bh, const __bf16* __restrict__ Bl,
    const float* __restrict__ bias, float* __restrict__ C, int N, int K) {
  __shared__ __attribute__((aligned(16))) __bf16 sAh[BM * LDA];
  __shared__ __attribute__((aligned(16))) __bf16 sAl[BM * LDA];
  __shared__ __attribute__((aligned(16))) __bf16 sBh[BN * LDB];
  __shared__ __attribute__((aligned(16))) __bf16 sBl[BN * LDB];

  const int tid = threadIdx.x;
  const int m0 = blockIdx.y * BM;
  const int n0 = blockIdx.x * BN;
  const int lane = tid & 31, wv = tid >> 5;
  const int khalf = lane >> 4, lrow = lane & 15;
  const int wm = wv >> 2, wn = wv & 3;

  // fixed per-thread staging slots (16B chunks of 8 bf16)
  const int arow_s = tid >> 2, acq = (tid & 3) * 8;            // A: 64 rows x 4 chunks
  const int brow0 = tid >> 2, brow1 = (tid + 256) >> 2;        // B: 128 rows x 4 chunks
  const int bcq = (tid & 3) * 8;
  const unsigned ldsAh = (unsigned)(uintptr_t)&sAh[arow_s * LDA + acq];
  const unsigned ldsAl = (unsigned)(uintptr_t)&sAl[arow_s * LDA + acq];
  const unsigned ldsBh0 = (unsigned)(uintptr_t)&sBh[brow0 * LDB + bcq];
  const unsigned ldsBh1 = (unsigned)(uintptr_t)&sBh[brow1 * LDB + bcq];
  const unsigned ldsBl0 = (unsigned)(uintptr_t)&sBl[brow0 * LDB + bcq];
  const unsigned ldsBl1 = (unsigned)(uintptr_t)&sBl[brow1 * LDB + bcq];

  v8f acc[2][2];
#pragma unroll
  for (int a = 0; a < 2; ++a)
#pragma unroll
    for (int b = 0; b < 2; ++b)
#pragma unroll
      for (int r = 0; r < 8; ++r) acc[a][b][r] = 0.0f;

  for (int kk = 0; kk < K; kk += BK) {
    // 6 async 16B copies per thread fill the whole 64x32 + 128x32 bf16 tiles
    async_copy16(ldsAh, Ah + (size_t)(m0 + arow_s) * K + kk + acq);
    async_copy16(ldsAl, Al + (size_t)(m0 + arow_s) * K + kk + acq);
    async_copy16(ldsBh0, Bh + (size_t)(n0 + brow0) * K + kk + bcq);
    async_copy16(ldsBh1, Bh + (size_t)(n0 + brow1) * K + kk + bcq);
    async_copy16(ldsBl0, Bl + (size_t)(n0 + brow0) * K + kk + bcq);
    async_copy16(ldsBl1, Bl + (size_t)(n0 + brow1) * K + kk + bcq);
    wait_async0();
    __syncthreads();

    v16bf bh[2], bl[2];
#pragma unroll
    for (int nt = 0; nt < 2; ++nt) {
      int brow = (32 * wn + 16 * nt + lrow);
      bh[nt] = frag_lds(&sBh[brow * LDB], khalf);
      bl[nt] = frag_lds(&sBl[brow * LDB], khalf);
    }
#pragma unroll
    for (int mt = 0; mt < 2; ++mt) {
      int arow = (32 * wm + 16 * mt + lrow);
      v16bf ah = frag_lds(&sAh[arow * LDA], khalf);
      v16bf al = frag_lds(&sAl[arow * LDA], khalf);
#pragma unroll
      for (int nt = 0; nt < 2; ++nt) {
        acc[mt][nt] = wmma_bf16(ah, bh[nt], acc[mt][nt]);
        acc[mt][nt] = wmma_bf16(ah, bl[nt], acc[mt][nt]);
        acc[mt][nt] = wmma_bf16(al, bh[nt], acc[mt][nt]);
      }
    }
    __syncthreads();
  }

#pragma unroll
  for (int mt = 0; mt < 2; ++mt)
#pragma unroll
    for (int nt = 0; nt < 2; ++nt) {
      int n = n0 + 32 * wn + 16 * nt + lrow;
      float bv = bias[n];
#pragma unroll
      for (int r = 0; r < 8; ++r) {
        int m = m0 + 32 * wm + 16 * mt + 8 * khalf + r;
        C[(size_t)m * N + n] = acc[mt][nt][r] + bv;
      }
    }
}

// f32 matrix -> contiguous bf16 hi/lo planes (one float4 per thread).
__global__ __launch_bounds__(256) void split_planes_kernel(
    const float* __restrict__ src, __bf16* __restrict__ hi,
    __bf16* __restrict__ lo, long n4) {
  long i = (long)blockIdx.x * 256 + threadIdx.x;
  if (i >= n4) return;
  float4 v = ((const float4*)src)[i];
  __bf16* ph = hi + 4 * i;
  __bf16* pl = lo + 4 * i;
  split2(v.x, ph[0], pl[0]); split2(v.y, ph[1], pl[1]);
  split2(v.z, ph[2], pl[2]); split2(v.w, ph[3], pl[3]);
}

// ---------------------------------------------------------------------------
// Persistent recurrence scan (unchanged from previous round): 16 WGs,
// W_hh split-bf16 resident in 264KB LDS, per-step grid barrier.
// ---------------------------------------------------------------------------
__global__ __launch_bounds__(256) void rnn_scan_kernel(
    const float* __restrict__ xin, const float* __restrict__ Whh,
    const float* __restrict__ bhh, const float* __restrict__ alpha,
    const float* __restrict__ h0, float* __restrict__ hs,
    int* __restrict__ cnt, int use_cluster) {
  constexpr int LDW = 1024 + 8;
  extern __shared__ __bf16 smem[];
  __bf16* sWh = smem;
  __bf16* sWl = smem + 64 * LDW;

  const int g = blockIdx.x;
  const int tid = threadIdx.x;
  const int lane = tid & 31, wv = tid >> 5;
  const int khalf = lane >> 4, lrow = lane & 15;
  const int wm = wv >> 2, wn = wv & 3;
  const int mrow = 16 * wm + lrow;
  const int nloc = 16 * wn + lrow;

#pragma unroll 4
  for (int j = 0; j < 64; ++j) {
    int idx = tid + 256 * j;
    int row = idx >> 8, kq = idx & 255;
    float4 v = *(const float4*)(Whh + (size_t)(64 * g + row) * 1024 + 4 * kq);
    __bf16* ph = &sWh[row * LDW + 4 * kq];
    __bf16* pl = &sWl[row * LDW + 4 * kq];
    split2(v.x, ph[0], pl[0]); split2(v.y, ph[1], pl[1]);
    split2(v.z, ph[2], pl[2]); split2(v.w, ph[3], pl[3]);
  }
  __syncthreads();

  const int barrier_target = (int)gridDim.x;

  for (int s = 0; s < 128; ++s) {
    const float* hp = (s == 0) ? h0 : (hs + (size_t)(s - 1) * 32 * 1024);
    const float* arow = hp + (size_t)mrow * 1024;

    v8f acc;
#pragma unroll
    for (int r = 0; r < 8; ++r) acc[r] = 0.0f;

#pragma unroll 2
    for (int kk = 0; kk < 1024; kk += 32) {
      v16bf ah, al;
      const float* pa = arow + kk + (khalf << 3);
#pragma unroll
      for (int i = 0; i < 8; ++i) {
        float f = pa[i];
        __bf16 h, l; split2(f, h, l);
        ah[i] = h; al[i] = l;
      }
#pragma unroll
      for (int i = 0; i < 8; ++i) {
        float f = pa[16 + i];
        __bf16 h, l; split2(f, h, l);
        ah[8 + i] = h; al[8 + i] = l;
      }
      v16bf bh = frag_lds(&sWh[nloc * LDW + kk], khalf);
      v16bf bl = frag_lds(&sWl[nloc * LDW + kk], khalf);
      acc = wmma_bf16(ah, bh, acc);
      acc = wmma_bf16(ah, bl, acc);
      acc = wmma_bf16(al, bh, acc);
    }

    const float a_s = alpha[s];
    const int n = 64 * g + nloc;
    const float bv = bhh[n];
    const float* xp = xin + (size_t)s * 32 * 1024;
    float* outp = hs + (size_t)s * 32 * 1024;
#pragma unroll
    for (int r = 0; r < 8; ++r) {
      int m = 8 * khalf + r;
      outp[(size_t)m * 1024 + n] = a_s * tanhf(acc[r] + xp[(size_t)m * 1024 + n] + bv);
    }

    __builtin_amdgcn_fence(__ATOMIC_RELEASE, "agent");
    __syncthreads();
    if (use_cluster) {
      __builtin_amdgcn_s_cluster_barrier();
    } else if (tid == 0) {
      __hip_atomic_fetch_add(&cnt[s], 1, __ATOMIC_RELAXED, __HIP_MEMORY_SCOPE_AGENT);
      while (__hip_atomic_load(&cnt[s], __ATOMIC_ACQUIRE, __HIP_MEMORY_SCOPE_AGENT) <
             barrier_target) {
        __builtin_amdgcn_s_sleep(1);
      }
    }
    __syncthreads();
    __builtin_amdgcn_fence(__ATOMIC_ACQUIRE, "agent");
  }
}

__global__ void zero_cnt_kernel(int* c) { c[threadIdx.x] = 0; }

// ---------------------------------------------------------------------------
extern "C" void kernel_launch(void* const* d_in, const int* in_sizes, int n_in,
                              void* d_out, int out_size, void* d_ws, size_t ws_size,
                              hipStream_t stream) {
  (void)in_sizes; (void)n_in; (void)out_size;
  const int*   tok   = (const int*)d_in[0];
  const float* h0    = (const float*)d_in[1];
  const float* embW  = (const float*)d_in[2];
  const float* Wih   = (const float*)d_in[3];
  const float* Whh   = (const float*)d_in[4];
  const float* bih   = (const float*)d_in[5];
  const float* bhh   = (const float*)d_in[6];
  const float* alpha = (const float*)d_in[7];
  const float* decW  = (const float*)d_in[8];
  const float* decb  = (const float*)d_in[9];
  float* out = (float*)d_out;

  char* ws = (char*)d_ws;
  float* xin = (float*)ws;                          // 16 MiB
  float* hs  = (float*)(ws + (size_t)(16u << 20));  // 16 MiB
  int*   cnt = (int*)(ws + (size_t)(32u << 20));    // 512 B
  // fast-path bf16 planes
  const size_t off_hsH = (size_t)34u << 20;
  const size_t off_hsL = off_hsH + (size_t)8u * 1024 * 1024;
  const size_t off_BH  = off_hsL + (size_t)8u * 1024 * 1024;
  const size_t off_BL  = off_BH + (size_t)32000 * 1024 * 2;
  const size_t need    = off_BL + (size_t)32000 * 1024 * 2;
  __bf16* hsH = (__bf16*)(ws + off_hsH);
  __bf16* hsL = (__bf16*)(ws + off_hsL);
  __bf16* BH  = (__bf16*)(ws + off_BH);
  __bf16* BL  = (__bf16*)(ws + off_BL);

  zero_cnt_kernel<<<1, 128, 0, stream>>>(cnt);

  // encoder gather + input projection: xin = emb_W[tok] * W_ih^T + b_ih
  gemm_bias_kernel<true><<<dim3(1024 / BN, 4096 / BM), 256, 0, stream>>>(
      embW, tok, Wih, bih, xin, 1024, 1024);

  // persistent recurrence: one launch, 16-WG (cluster-sized) grid, 128 steps
  size_t smem = (size_t)2 * 64 * (1024 + 8) * 2;  // 264192 B < 320 KB/WGP
  rnn_scan_kernel<<<16, 256, smem, stream>>>(xin, Whh, bhh, alpha, h0, hs, cnt,
                                             /*use_cluster=*/0);

  // hidden_f = hs[127]
  (void)hipMemcpyAsync(out + (size_t)128 * 32 * 32000, hs + (size_t)127 * 32 * 1024,
                       (size_t)32 * 1024 * sizeof(float), hipMemcpyDeviceToDevice,
                       stream);

  if (ws_size >= need) {
    // pre-split once; decoder GEMM then streams pure bf16 via async->LDS
    split_planes_kernel<<<(unsigned)(4096 * 1024 / 4 / 256), 256, 0, stream>>>(
        hs, hsH, hsL, (long)4096 * 1024 / 4);
    split_planes_kernel<<<(unsigned)((size_t)32000 * 1024 / 4 / 256), 256, 0, stream>>>(
        decW, BH, BL, (long)32000 * 1024 / 4);
    gemm_bias_bf16_kernel<<<dim3(32000 / BN, 4096 / BM), 256, 0, stream>>>(
        hsH, hsL, BH, BL, decb, out, 32000, 1024);
  } else {
    // fallback: fused-conversion decoder
    gemm_bias_kernel<false><<<dim3(32000 / BN, 4096 / BM), 256, 0, stream>>>(
        hs, nullptr, decW, decb, out, 32000, 1024);
  }
}
